// myLSTMCell_group2_54571854463575
// MI455X (gfx1250) — compile-verified, matrix operand order checked by reading, not run on hardware
//
#include <hip/hip_runtime.h>
#include <hip/hip_bf16.h>
#include <math.h>

// ---------------------------------------------------------------------------
// myLSTMCell_group2 for MI455X (gfx1250): bf16 WMMA pipeline + TDM staging.
//   B=8192, IN=1024, H=1024, WRANK=256, UR0=64, UR1=32, G=2, HG=512
// Phase 0: pack all weights into per-lane v_wmma bf16 B-fragment layout.
//          Gate weights are grouped so each gates-block's 88KB B working set
//          is one contiguous chunk -> single 1-D tensor_load_to_lds.
// Phase 1: P[B,448] = [x@W | hg0@U0 | hg1@U1 | hg1@UU0 | hg0@UU1]  (bf16)
// Phase 2: 4-gate GEMM (M=8192,N=1024,K=352), B-fragments from LDS (TDM),
//          fused LSTM epilogue.
// ---------------------------------------------------------------------------

typedef __attribute__((ext_vector_type(16))) __bf16 v16bf;
typedef __attribute__((ext_vector_type(8)))  __bf16 v8bf;
typedef __attribute__((ext_vector_type(8)))  float  v8f;
typedef unsigned int u32;
typedef __attribute__((ext_vector_type(4))) u32 u32x4;
typedef __attribute__((ext_vector_type(8))) int  i32x8;
typedef __attribute__((ext_vector_type(4))) int  i32x4;

#define B_DIM   8192
#define IN_DIM  1024
#define H_DIM   1024
#define WRANK   256
#define UR0     64
#define UR1     32
#define HG      512
#define PCOLS   448          // 256 + 2*64 + 2*32
#define KT_G    11           // ceil(352/32): 8 (xw) + 2 (U) + 1 (UU)

// per-gates-block contiguous B chunk: 4 gates * 11 kt * 2 ntiles * 1KB
#define CHUNK_FRAGS   (4 * KT_G * 2)     // 88
#define CHUNK_HALVES  (CHUNK_FRAGS * 512)  // 45056
#define CHUNK_BYTES   (CHUNK_HALVES * 2)   // 90112

// workspace layout (bytes, all 256B aligned)
#define WS_P     0u                         // 8192*448*2    = 7,340,032
#define WS_PW    7340032u                   // 1024*256*2    =   524,288
#define WS_PU    7864320u                   // 2*512*64*2    =   131,072
#define WS_PUU   7995392u                   // 2*512*32*2    =    65,536
#define WS_PG    8060928u                   // 4*2*352*512*2 = 2,883,584

#if defined(__has_builtin)
#if __has_builtin(__builtin_amdgcn_tensor_load_to_lds)
#define HAVE_TDM 1
#endif
#endif
#ifndef HAVE_TDM
#define HAVE_TDM 0
#endif

// fragment element -> (k,n) decode shared by all pack kernels:
//   lane = within/16, idx = within%16
//   n = nt*16 + (lane&15)
//   k = kt*32 + (lane>>4)*8 + (idx&7) + ((idx>>3)<<4)
__device__ __forceinline__ void frag_kn(int within, int kt, int nt, int& k, int& n) {
    int lane = within >> 4, idx = within & 15;
    n = nt * 16 + (lane & 15);
    k = kt * 32 + ((lane >> 4) << 3) + (idx & 7) + ((idx >> 3) << 4);
}

// ---- pack W [1024,256] : frag = kt*16+nt, kt<32, nt<16 ------------------
__global__ __launch_bounds__(256) void pack_w(const float* __restrict__ W,
                                              __bf16* __restrict__ out) {
    int e = blockIdx.x * 256 + threadIdx.x;
    if (e >= 32 * 16 * 512) return;
    int within = e & 511, frag = e >> 9;
    int nt = frag & 15, kt = frag >> 4;
    int k, n; frag_kn(within, kt, nt, k, n);
    out[e] = (__bf16)W[k * WRANK + n];
}

// ---- pack U [2,512,64] : frag = g*64 + kt*4 + nt, kt<16, nt<4 -----------
__global__ __launch_bounds__(256) void pack_u(const float* __restrict__ U,
                                              __bf16* __restrict__ out) {
    int e = blockIdx.x * 256 + threadIdx.x;
    if (e >= 2 * 16 * 4 * 512) return;
    int within = e & 511, frag = e >> 9;
    int nt = frag & 3, kt = (frag >> 2) & 15, g = frag >> 6;
    int k, n; frag_kn(within, kt, nt, k, n);
    out[e] = (__bf16)U[(g * HG + k) * UR0 + n];
}

// ---- pack UU [2,512,32] : frag = g*32 + kt*2 + nt, kt<16, nt<2 ----------
__global__ __launch_bounds__(256) void pack_uu(const float* __restrict__ UU,
                                               __bf16* __restrict__ out) {
    int e = blockIdx.x * 256 + threadIdx.x;
    if (e >= 2 * 16 * 2 * 512) return;
    int within = e & 511, frag = e >> 9;
    int nt = frag & 1, kt = (frag >> 1) & 15, g = frag >> 5;
    int k, n; frag_kn(within, kt, nt, k, n);
    out[e] = (__bf16)UU[(g * HG + k) * UR1 + n];
}

// ---- pack gate weights, TDM-chunked layout -------------------------------
// frag = (((g*16 + ntp)*4 + q)*11 + kt)*2 + sp ; column n = (ntp*2+sp)*16+...
// => chunk for (g,ntp) is contiguous: 88 frags = 88KB.
__global__ __launch_bounds__(256) void pack_g(
    const float* __restrict__ W1, const float* __restrict__ W2,
    const float* __restrict__ W3, const float* __restrict__ W4,
    const float* __restrict__ U1, const float* __restrict__ U2,
    const float* __restrict__ U3, const float* __restrict__ U4,
    const float* __restrict__ UU1, const float* __restrict__ UU2,
    const float* __restrict__ UU3, const float* __restrict__ UU4,
    __bf16* __restrict__ out) {
    int e = blockIdx.x * 256 + threadIdx.x;
    if (e >= 4 * 2 * KT_G * 32 * 512) return;
    int within = e & 511, frag = e >> 9;
    int sp  = frag & 1;
    int r1  = frag >> 1;
    int kt  = r1 % KT_G;
    int r2  = r1 / KT_G;
    int q   = r2 & 3;
    int r3  = r2 >> 2;
    int ntp = r3 & 15;
    int g   = r3 >> 4;
    int nt  = ntp * 2 + sp;
    int k, n; frag_kn(within, kt, nt, k, n);
    const float* Wq[4]  = {W1, W2, W3, W4};
    const float* Uq[4]  = {U1, U2, U3, U4};
    const float* UUq[4] = {UU1, UU2, UU3, UU4};
    float v;
    if (k < WRANK)            v = Wq[q][k * H_DIM + g * HG + n];
    else if (k < WRANK + UR0) v = Uq[q][(g * UR0 + (k - WRANK)) * HG + n];
    else                      v = UUq[q][(g * UR1 + (k - WRANK - UR0)) * HG + n];
    out[e] = (__bf16)v;
}

// ---------------------------------------------------------------------------
// Phase 1: projections P[B,448]. grid (64, 28); 8 waves/block stacked in M.
// ---------------------------------------------------------------------------
__global__ __launch_bounds__(256) void proj_kernel(
    const float* __restrict__ x, const float* __restrict__ h,
    const __bf16* __restrict__ pW, const __bf16* __restrict__ pU,
    const __bf16* __restrict__ pUU, __bf16* __restrict__ P) {
    const int wave = threadIdx.x >> 5, lane = threadIdx.x & 31;
    const int l16 = lane & 15, hif = lane >> 4;
    const int m0 = blockIdx.x * 128 + wave * 16;
    const int nt = blockIdx.y;

    const float* Abase; int ldA, ktiles, ntFrags, sub, pcol;
    const __bf16* Bp;
    if (nt < 16) {
        Abase = x; ldA = IN_DIM; ktiles = 32;
        Bp = pW; ntFrags = 16; sub = nt; pcol = nt * 16;
    } else if (nt < 24) {
        int t = nt - 16, g = t >> 2; sub = t & 3;
        Abase = h + g * HG; ldA = H_DIM; ktiles = 16;
        Bp = pU + (size_t)g * (16 * 4 * 512); ntFrags = 4;
        pcol = 256 + g * 64 + sub * 16;
    } else {
        int t = nt - 24, g = t >> 1; sub = t & 1;
        Abase = h + (g ^ 1) * HG; ldA = H_DIM; ktiles = 16;   // group rotation
        Bp = pUU + (size_t)g * (16 * 2 * 512); ntFrags = 2;
        pcol = 384 + g * 32 + sub * 16;
    }

    const float* arow = Abase + (size_t)(m0 + l16) * ldA;
    v8f acc = {};
    for (int kt = 0; kt < ktiles; ++kt) {
        const int kb = kt * 32 + hif * 8;
        const float4 a0 = *(const float4*)(arow + kb);
        const float4 a1 = *(const float4*)(arow + kb + 4);
        const float4 a2 = *(const float4*)(arow + kb + 16);
        const float4 a3 = *(const float4*)(arow + kb + 20);
        v16bf a;
        a[0] = (__bf16)a0.x;  a[1] = (__bf16)a0.y;  a[2]  = (__bf16)a0.z;  a[3]  = (__bf16)a0.w;
        a[4] = (__bf16)a1.x;  a[5] = (__bf16)a1.y;  a[6]  = (__bf16)a1.z;  a[7]  = (__bf16)a1.w;
        a[8] = (__bf16)a2.x;  a[9] = (__bf16)a2.y;  a[10] = (__bf16)a2.z;  a[11] = (__bf16)a2.w;
        a[12] = (__bf16)a3.x; a[13] = (__bf16)a3.y; a[14] = (__bf16)a3.z;  a[15] = (__bf16)a3.w;
        const v16bf b = *(const v16bf*)(Bp + (((size_t)(kt * ntFrags + sub)) << 9) + lane * 16);
        acc = __builtin_amdgcn_wmma_f32_16x16x32_bf16(false, a, false, b,
                                                      (short)0, acc, false, false);
    }
    const int n = pcol + l16;
#pragma unroll
    for (int r = 0; r < 8; ++r) {
        const int m = m0 + r + hif * 8;
        P[(size_t)m * PCOLS + n] = (__bf16)acc[r];
    }
}

// ---------------------------------------------------------------------------
// Phase 2: 4-gate GEMM (K=352) + fused LSTM epilogue. grid (64, 32).
// Per block: TDM-stage 88KB of pre-swizzled B fragments into LDS once
// (wave 0 issues tensor_load_to_lds, waits TENSORcnt, barrier), then all
// 8 waves read fragments via ds_load_b128. Each wave: 16(M) x 32(N) x 4 gates.
// ---------------------------------------------------------------------------
__global__ __launch_bounds__(256) void gates_kernel(
    const __bf16* __restrict__ P, const __bf16* __restrict__ pG,
    const float* __restrict__ cin,
    const float* __restrict__ bias_i, const float* __restrict__ bias_f,
    const float* __restrict__ bias_c, const float* __restrict__ bias_o,
    float* __restrict__ c_next, float* __restrict__ h_next) {
    extern __shared__ __bf16 smem[];            // CHUNK_BYTES dynamic LDS

    const int wave = threadIdx.x >> 5, lane = threadIdx.x & 31;
    const int l16 = lane & 15, hif = lane >> 4;
    const int m0 = blockIdx.x * 128 + wave * 16;
    const int jbase = blockIdx.y * 32;
    const int g = jbase >> 9;                   // column group (512 cols each)
    const int ntp = (jbase & 511) >> 5;         // 32-column pair index in group
    const int chunk = g * 16 + ntp;

#if HAVE_TDM
    if (wave == 0) {
        const unsigned long long ga =
            (unsigned long long)(uintptr_t)pG + (unsigned long long)chunk * CHUNK_BYTES;
        // D# group 0: count=1 | lds_addr=0 | global_addr | type=2
        u32x4 g0;
        g0[0] = 1u;
        g0[1] = 0u;
        g0[2] = (u32)(ga & 0xffffffffu);
        g0[3] = (u32)((ga >> 32) & 0x01ffffffu) | (2u << 30);
        // D# group 1: data_size=8B, 1-D tile of 11264 elements (=90112 B)
        i32x8 g1;
        g1[0] = (int)(3u << 16);                 // data_size = 3 (8 bytes)
        g1[1] = (int)(0x2C00u << 16);            // tensor_dim0[15:0] = 11264
        g1[2] = (int)(1u << 16);                 // tensor_dim0 hi=0, tensor_dim1=1
        g1[3] = (int)(0x2C00u << 16);            // tile_dim0 = 11264
        g1[4] = 1;                               // tile_dim1 = 1
        g1[5] = 0x2C00;                          // tensor_dim0_stride = 11264
        g1[6] = 0;
        g1[7] = 0;
        i32x4 gz = {0, 0, 0, 0};
#if __clang_major__ >= 23
        i32x8 gz8 = {0, 0, 0, 0, 0, 0, 0, 0};
        __builtin_amdgcn_tensor_load_to_lds(g0, g1, gz, gz, gz8, 0);
#else
        __builtin_amdgcn_tensor_load_to_lds(g0, g1, gz, gz, 0);
#endif
        __builtin_amdgcn_s_wait_tensorcnt(0);
    }
    __syncthreads();
#else
    {   // cooperative fallback: global -> LDS copy, 16B per thread per step
        const uint4* src = (const uint4*)((const char*)pG + (size_t)chunk * CHUNK_BYTES);
        uint4* dst = (uint4*)smem;
        for (int i = threadIdx.x; i < CHUNK_BYTES / 16; i += 256) dst[i] = src[i];
        __syncthreads();
    }
#endif

    v8f acc[4][2];
#pragma unroll
    for (int q = 0; q < 4; ++q) { acc[q][0] = {}; acc[q][1] = {}; }

    const __bf16* arow = P + (size_t)(m0 + l16) * PCOLS;
    for (int kt = 0; kt < KT_G; ++kt) {
        const int k0 = kt * 32;
        int pcb;                                // map K in [0,352) -> P column
        if (k0 < WRANK)            pcb = k0;
        else if (k0 < WRANK + UR0) pcb = 256 + g * 64 + (k0 - WRANK);
        else                       pcb = 384 + g * 32 + (k0 - WRANK - UR0);
        const int cb = pcb + hif * 8;
        const v8bf alo = *(const v8bf*)(arow + cb);
        const v8bf ahi = *(const v8bf*)(arow + cb + 16);
        const v16bf a = __builtin_shufflevector(alo, ahi,
            0, 1, 2, 3, 4, 5, 6, 7, 8, 9, 10, 11, 12, 13, 14, 15);
#pragma unroll
        for (int q = 0; q < 4; ++q) {
#pragma unroll
            for (int s = 0; s < 2; ++s) {
                const int lf = (q * KT_G + kt) * 2 + s;     // local frag in chunk
                const v16bf b = *(const v16bf*)(smem + lf * 512 + lane * 16);
                acc[q][s] = __builtin_amdgcn_wmma_f32_16x16x32_bf16(
                    false, a, false, b, (short)0, acc[q][s], false, false);
            }
        }
    }

#pragma unroll
    for (int s = 0; s < 2; ++s) {
        const int n = jbase + s * 16 + l16;
        const float bi = bias_i[n], bf = bias_f[n], bo = bias_o[n], bc = bias_c[n];
#pragma unroll
        for (int r = 0; r < 8; ++r) {
            const int m = m0 + r + hif * 8;
            const float vi = acc[0][s][r] + bi;
            const float vf = acc[1][s][r] + bf;
            const float vo = acc[2][s][r] + bo;
            const float vc = acc[3][s][r] + bc;
            const float iG = 1.f / (1.f + __expf(-vi));
            const float fG = 1.f / (1.f + __expf(-vf));
            const float oG = 1.f / (1.f + __expf(-vo));
            const float ct = tanhf(vc);
            const float cp = cin[(size_t)m * H_DIM + n];
            const float cn = fG * cp + iG * ct;
            const float hn = oG * tanhf(cn);
            c_next[(size_t)m * H_DIM + n] = cn;
            h_next[(size_t)m * H_DIM + n] = hn;
        }
    }
}

// ---------------------------------------------------------------------------
extern "C" void kernel_launch(void* const* d_in, const int* in_sizes, int n_in,
                              void* d_out, int out_size, void* d_ws, size_t ws_size,
                              hipStream_t stream) {
    const float* x  = (const float*)d_in[0];
    const float* h  = (const float*)d_in[1];
    const float* c  = (const float*)d_in[2];
    const float* W  = (const float*)d_in[3];
    const float* W1 = (const float*)d_in[4];
    const float* W2 = (const float*)d_in[5];
    const float* W3 = (const float*)d_in[6];
    const float* W4 = (const float*)d_in[7];
    const float* U  = (const float*)d_in[8];
    const float* U1 = (const float*)d_in[9];
    const float* U2 = (const float*)d_in[10];
    const float* U3 = (const float*)d_in[11];
    const float* U4 = (const float*)d_in[12];
    const float* UU  = (const float*)d_in[13];
    const float* UU1 = (const float*)d_in[14];
    const float* UU2 = (const float*)d_in[15];
    const float* UU3 = (const float*)d_in[16];
    const float* UU4 = (const float*)d_in[17];
    const float* bias_f = (const float*)d_in[18];
    const float* bias_i = (const float*)d_in[19];
    const float* bias_c = (const float*)d_in[20];
    const float* bias_o = (const float*)d_in[21];

    char* ws = (char*)d_ws;
    __bf16* P   = (__bf16*)(ws + WS_P);
    __bf16* pW  = (__bf16*)(ws + WS_PW);
    __bf16* pU  = (__bf16*)(ws + WS_PU);
    __bf16* pUU = (__bf16*)(ws + WS_PUU);
    __bf16* pG  = (__bf16*)(ws + WS_PG);

    pack_w <<<(32 * 16 * 512) / 256, 256, 0, stream>>>(W, pW);
    pack_u <<<(2 * 16 * 4 * 512) / 256, 256, 0, stream>>>(U, pU);
    pack_uu<<<(2 * 16 * 2 * 512) / 256, 256, 0, stream>>>(UU, pUU);
    pack_g <<<(4 * 2 * KT_G * 32 * 512) / 256, 256, 0, stream>>>(
        W1, W2, W3, W4, U1, U2, U3, U4, UU1, UU2, UU3, UU4, pG);

    proj_kernel<<<dim3(B_DIM / 128, 28), 256, 0, stream>>>(x, h, pW, pU, pUU, P);

    float* out = (float*)d_out;
    gates_kernel<<<dim3(B_DIM / 128, H_DIM / 32), 256, CHUNK_BYTES, stream>>>(
        P, pG, c, bias_i, bias_f, bias_c, bias_o,
        out, out + (size_t)B_DIM * H_DIM);
}
